// PhysicsInformedBKLayer_82927228551615
// MI455X (gfx1250) — compile-verified, matrix operand order checked by reading, not run on hardware
//
#include <hip/hip_runtime.h>
#include <hip/hip_bf16.h>

typedef __attribute__((ext_vector_type(16))) __bf16 v16bf;
typedef __attribute__((ext_vector_type(8)))  float  v8f;
typedef __attribute__((ext_vector_type(4)))  unsigned v4u;
typedef __attribute__((ext_vector_type(8)))  unsigned v8u;

#define TB   4
#define TN   2048
#define TD   512
#define TE   4
#define TH   2048
#define TTOK (TB * TN)   // 8192 tokens

// ---------------- workspace layout (bytes) ----------------
static const size_t OFF_W1P  = 0;
static const size_t OFF_W2P  = OFF_W1P  + (size_t)TE * TD * TH * 2;
static const size_t OFF_XN   = OFF_W2P  + (size_t)TE * TH * TD * 2;
static const size_t OFF_MOE  = OFF_XN   + (size_t)TTOK * TD * 2;
static const size_t OFF_TP   = OFF_MOE  + (size_t)TTOK * TD * 4;
static const size_t OFF_V    = OFF_TP   + (size_t)TTOK * 4;
static const size_t OFF_U    = OFF_V    + (size_t)TTOK * 4;
static const size_t OFF_SPEC = OFF_U    + (size_t)TTOK * 8;
static const size_t OFF_LIST = OFF_SPEC + (size_t)TTOK * 8;
static const size_t OFF_CNT  = OFF_LIST + (size_t)TE * TTOK * 4;

// k index inside a 16-bit A/B fragment: lane-group g (0/1), element j (0..15)
__device__ __forceinline__ int frag_k(int g, int j) {
    int p = j >> 1;
    int base = (p < 4) ? (g * 8 + 2 * p) : (16 + g * 8 + 2 * (p - 4));
    return base + (j & 1);
}

// branch-free tanh via v_exp_f32
__device__ __forceinline__ float fast_tanh(float z) {
    float az = fabsf(z);
    float e = __expf(-2.0f * az);
    float t = (1.0f - e) / (1.0f + e);
    return copysignf(t, z);
}

// ---------------- kernel: zero expert counters ----------------
__global__ void zero_counts_kernel(int* counts) {
    if (threadIdx.x < TE) counts[threadIdx.x] = 0;
}

// ---------------- kernel: pack w1 (D x H per expert) into B-fragment order ----------------
// dest layout: [e][kt=16][nt=128][512 bf16]  (K tile = 32 over D, N tile = 16 over H)
__global__ void pack_w1_kernel(const float* __restrict__ w1, __bf16* __restrict__ w1p) {
    int t = blockIdx.x * blockDim.x + threadIdx.x;
    if (t >= TE * TD * TH) return;
    int inner = t & 511;
    int j = inner & 15;
    int L = inner >> 4;
    int rest = t >> 9;
    int nt = rest & 127; rest >>= 7;
    int kt = rest & 15;
    int e  = rest >> 4;
    int n = L & 15, g = L >> 4;
    int k = frag_k(g, j);
    float src = w1[((size_t)e * TD + (kt * 32 + k)) * TH + nt * 16 + n];
    w1p[t] = (__bf16)src;
}

// ---------------- kernel: pack w2 (H x D per expert) into B-fragment order ----------------
// dest layout: [e][kt=64][nt=32][512 bf16]  (K tile = 32 over H, N tile = 16 over D)
__global__ void pack_w2_kernel(const float* __restrict__ w2, __bf16* __restrict__ w2p) {
    int t = blockIdx.x * blockDim.x + threadIdx.x;
    if (t >= TE * TH * TD) return;
    int inner = t & 511;
    int j = inner & 15;
    int L = inner >> 4;
    int rest = t >> 9;
    int nt = rest & 31; rest >>= 5;
    int kt = rest & 63;
    int e  = rest >> 6;
    int n = L & 15, g = L >> 4;
    int k = frag_k(g, j);
    float src = w2[((size_t)e * TH + (kt * 32 + k)) * TD + nt * 16 + n];
    w2p[t] = (__bf16)src;
}

// ---------------- kernel: layernorm + gate + bucket by expert ----------------
__global__ __launch_bounds__(256) void ln_gate_kernel(
    const float* __restrict__ x, const float* __restrict__ gamma, const float* __restrict__ beta,
    const float* __restrict__ gate_w, const float* __restrict__ gate_b,
    __bf16* __restrict__ xn_out, float* __restrict__ tp_out,
    int* __restrict__ lists, int* __restrict__ counts)
{
    int wave = threadIdx.x >> 5;
    int lane = threadIdx.x & 31;
    int token = blockIdx.x * 8 + wave;
    if (token >= TTOK) return;

    const float* xr = x + (size_t)token * TD;
    float vals[16];
    float s = 0.f, s2 = 0.f;
#pragma unroll
    for (int i = 0; i < 16; i++) {
        float v = xr[i * 32 + lane];
        vals[i] = v; s += v; s2 += v * v;
    }
#pragma unroll
    for (int m = 16; m >= 1; m >>= 1) { s += __shfl_xor(s, m, 32); s2 += __shfl_xor(s2, m, 32); }
    float mu = s * (1.f / TD);
    float var = s2 * (1.f / TD) - mu * mu;
    float rstd = rsqrtf(var + 1e-5f);

    float l[TE] = {0.f, 0.f, 0.f, 0.f};
#pragma unroll
    for (int i = 0; i < 16; i++) {
        int d = i * 32 + lane;
        float xnv = (vals[i] - mu) * rstd * gamma[d] + beta[d];
        xn_out[(size_t)token * TD + d] = (__bf16)xnv;
        const float* gw = gate_w + (size_t)d * TE;
#pragma unroll
        for (int e = 0; e < TE; e++) l[e] += xnv * gw[e];
    }
#pragma unroll
    for (int e = 0; e < TE; e++) {
#pragma unroll
        for (int m = 16; m >= 1; m >>= 1) l[e] += __shfl_xor(l[e], m, 32);
        l[e] += gate_b[e];
    }
    float mx = fmaxf(fmaxf(l[0], l[1]), fmaxf(l[2], l[3]));
    float ex[TE]; float se = 0.f;
#pragma unroll
    for (int e = 0; e < TE; e++) { ex[e] = __expf(l[e] - mx); se += ex[e]; }
    int idx = 0; float best = l[0];
#pragma unroll
    for (int e = 1; e < TE; e++) if (l[e] > best) { best = l[e]; idx = e; }
    float tp = ex[idx] / se;

    if (lane == 0) {
        tp_out[token] = tp;
        int slot = atomicAdd(&counts[idx], 1);
        lists[idx * TTOK + slot] = token;
    }
}

// load a 16x32 bf16 A-fragment from an LDS row (row base passed by caller)
__device__ __forceinline__ v16bf load_afrag(const __bf16* row, int k0, int g) {
    union { v16bf v; unsigned u[8]; } f;
#pragma unroll
    for (int p = 0; p < 8; p++) {
        int kb = (p < 4) ? (g * 8 + 2 * p) : (16 + g * 8 + 2 * (p - 4));
        f.u[p] = *(const unsigned*)(row + k0 + kb);
    }
    return f.v;
}

// ---------------- kernel: MoE double GEMM via WMMA + TDM gather of xn rows ----------------
// grid: (tiles=TTOK/16, experts=E); block 256 = 8 waves; 16 same-expert tokens per block
// dynamic LDS: [0,16384) xn tile (16x512 bf16), [16384,32768) h tile, then tok/tp arrays
__global__ __launch_bounds__(256) void moe_gemm_kernel(
    const __bf16* __restrict__ xn, const __bf16* __restrict__ w1p, const __bf16* __restrict__ w2p,
    const float* __restrict__ b1, const float* __restrict__ b2,
    const float* __restrict__ tp, const int* __restrict__ lists, const int* __restrict__ counts,
    float* __restrict__ moe)
{
    extern __shared__ __align__(16) char smem[];
    __bf16* s_xn  = (__bf16*)smem;              // 16 x 512 bf16, LDS offset 0
    __bf16* s_h   = (__bf16*)(smem + 16384);    // 16 x 512 bf16
    int*    s_tok = (int*)  (smem + 32768);     // 16 ints
    float*  s_tp  = (float*)(smem + 32832);     // 16 floats

    int e = blockIdx.y;
    int cnt = counts[e];
    int base = blockIdx.x * 16;
    if (base >= cnt) return;
    int valid = cnt - base; if (valid > 16) valid = 16;

    int tid = threadIdx.x;
    if (tid < 16) {
        int li = base + tid; if (li > cnt - 1) li = cnt - 1;
        int tok = lists[e * TTOK + li];
        s_tok[tid] = tok;
        s_tp[tid] = tp[tok];
    }
    __syncthreads();

    int wave = tid >> 5, lane = tid & 31;

    // ---- TDM gather: 16 token rows (512 bf16 each) from xn -> LDS offset 0 ----
    if (wave == 0) {
        unsigned ti[16];
#pragma unroll
        for (int i = 0; i < 16; i++)
            ti[i] = (unsigned)__builtin_amdgcn_readfirstlane(s_tok[i]) & 0xFFFFu;

        unsigned long long ga = (unsigned long long)(uintptr_t)xn;
        // D# group0: count=1, gather_mode=1 (16-bit idx), lds_addr=0, global_addr, type=2
        v4u g0 = { 1u | (1u << 31),
                   0u,
                   (unsigned)(ga & 0xFFFFFFFFu),
                   (unsigned)((ga >> 32) & 0x1FFFFFFu) | (2u << 30) };
        // D# group1: data_size=2B; tensor_dim0=512; tensor_dim1=8192 rows;
        //            tile_dim0=512; tile_dim1=16 indices; dim0 stride=512
        v8u g1 = { (1u << 16),
                   (512u << 16),
                   ((unsigned)TTOK << 16),
                   (512u << 16),
                   16u,
                   512u,
                   0u, 0u };
        // D# groups 2-3: sixteen 16-bit row indices
        v4u g2 = { ti[0] | (ti[1] << 16),  ti[2] | (ti[3] << 16),
                   ti[4] | (ti[5] << 16),  ti[6] | (ti[7] << 16) };
        v4u g3 = { ti[8] | (ti[9] << 16),  ti[10] | (ti[11] << 16),
                   ti[12] | (ti[13] << 16), ti[14] | (ti[15] << 16) };
        asm volatile("tensor_load_to_lds %0, %1, %2, %3"
                     :: "s"(g0), "s"(g1), "s"(g2), "s"(g3)
                     : "memory");
        __builtin_amdgcn_s_wait_tensorcnt(0);
    }
    __syncthreads();

    int lm = lane & 15, g = lane >> 4;

    v8f acc2[4];
#pragma unroll
    for (int j = 0; j < 4; j++)
#pragma unroll
        for (int q = 0; q < 8; q++) acc2[j][q] = 0.f;

    // H processed in 4 chunks of 512 columns
    for (int c = 0; c < 4; c++) {
        v8f acc1[4];
#pragma unroll
        for (int j = 0; j < 4; j++)
#pragma unroll
            for (int q = 0; q < 8; q++) acc1[j][q] = 0.f;

        // ---- GEMM1: (16 x 512) * (512 x 512-chunk) ----
        for (int ks = 0; ks < 16; ks++) {
            v16bf A = load_afrag(s_xn + lm * 512, ks * 32, g);
#pragma unroll
            for (int j = 0; j < 4; j++) {
                int ht = c * 32 + wave * 4 + j;            // global h n-tile (0..127)
                const v16bf* bp = (const v16bf*)(w1p +
                    (((size_t)e * 16 + ks) * 128 + ht) * 512 + lane * 16);
                v16bf Bf = *bp;
                acc1[j] = __builtin_amdgcn_wmma_f32_16x16x32_bf16(
                    false, A, false, Bf, (short)0, acc1[j], false, false);
            }
        }
        // bias + gelu(tanh approx, branch-free) -> bf16 into LDS h tile
#pragma unroll
        for (int j = 0; j < 4; j++) {
            int colbase = (wave * 4 + j) * 16;             // within-chunk col tile
            int hglob = c * 512 + colbase + lm;            // global h column
            float bn = b1[(size_t)e * TH + hglob];
#pragma unroll
            for (int q = 0; q < 8; q++) {
                int m = q + 8 * g;
                float val = acc1[j][q] + bn;
                float t3 = val + 0.044715f * val * val * val;
                float gel = 0.5f * val * (1.f + fast_tanh(0.7978845608028654f * t3));
                s_h[m * 512 + colbase + lm] = (__bf16)gel;
            }
        }
        __syncthreads();

        // ---- GEMM2 partial: (16 x 512-chunk) * (512-chunk x 512) ----
        for (int ks = 0; ks < 16; ks++) {
            v16bf A = load_afrag(s_h + lm * 512, ks * 32, g);
#pragma unroll
            for (int j = 0; j < 4; j++) {
                int nt2 = wave * 4 + j;                    // output d n-tile (0..31)
                int ktg = c * 16 + ks;                     // global k-tile over H (0..63)
                const v16bf* bp = (const v16bf*)(w2p +
                    (((size_t)e * 64 + ktg) * 32 + nt2) * 512 + lane * 16);
                v16bf Bf = *bp;
                acc2[j] = __builtin_amdgcn_wmma_f32_16x16x32_bf16(
                    false, A, false, Bf, (short)0, acc2[j], false, false);
            }
        }
        __syncthreads();
    }

    // finalize: + b2, * top_p, scatter to moe
#pragma unroll
    for (int j = 0; j < 4; j++) {
        int col = (wave * 4 + j) * 16 + lm;
        float bb = b2[(size_t)e * TD + col];
#pragma unroll
        for (int q = 0; q < 8; q++) {
            int m = q + 8 * g;
            if (m < valid) {
                float val = (acc2[j][q] + bb) * s_tp[m];
                moe[(size_t)s_tok[m] * TD + col] = val;
            }
        }
    }
}

// ---------------- kernel: v = clip(moe . proj_w + proj_b, +-3) ----------------
__global__ __launch_bounds__(256) void proj_v_kernel(
    const float* __restrict__ moe, const float* __restrict__ proj_w,
    const float* __restrict__ proj_b, float* __restrict__ v)
{
    int wave = threadIdx.x >> 5;
    int lane = threadIdx.x & 31;
    int token = blockIdx.x * 8 + wave;
    if (token >= TTOK) return;
    const float* mr = moe + (size_t)token * TD;
    float s = 0.f;
#pragma unroll
    for (int i = 0; i < 16; i++) {
        int d = i * 32 + lane;
        s += mr[d] * proj_w[d];
    }
#pragma unroll
    for (int m = 16; m >= 1; m >>= 1) s += __shfl_xor(s, m, 32);
    if (lane == 0) {
        float val = s + proj_b[0];
        val = fminf(3.0f, fmaxf(-3.0f, val));
        v[token] = val;
    }
}

// ---------------- kernel: Green's-function diagonal scan (sequential along N) ----------------
__global__ void green_scan_kernel(const float* __restrict__ v,
                                  float* __restrict__ u, float* __restrict__ spec)
{
    int b = threadIdx.x;
    if (b >= TB) return;
    float ur = 0.f, ui = 0.f;
    for (int i = 0; i < TN; i++) {
        size_t t = (size_t)b * TN + i;
        float ar = 2.0f - v[t];
        float ai = 1.0f;
        if (i == 0) { ur = ar; ui = ai; }
        else {
            float den = ur * ur + ui * ui;
            float inv_r = ur / den, inv_i = -ui / den;
            ur = ar - inv_r; ui = ai - inv_i;
        }
        u[t * 2] = ur; u[t * 2 + 1] = ui;
    }
    float wr = 0.f, wi = 0.f;
    for (int i = TN - 1; i >= 0; i--) {
        size_t t = (size_t)b * TN + i;
        float ar = 2.0f - v[t];
        float ai = 1.0f;
        if (i == TN - 1) { wr = ar; wi = ai; }
        else {
            float den = wr * wr + wi * wi;
            float inv_r = wr / den, inv_i = -wi / den;
            wr = ar - inv_r; wi = ai - inv_i;
        }
        float dr = u[t * 2] + wr - ar;
        float di = u[t * 2 + 1] + wi - ai;
        float den = dr * dr + di * di;
        float gr = dr / den, gi = -di / den;
        gr = fminf(10.0f, fmaxf(-10.0f, gr));
        gi = fminf(10.0f, fmaxf(-10.0f, gi));
        spec[t * 2] = gr; spec[t * 2 + 1] = gi;
    }
}

// ---------------- kernel: out = moe + bk * (spec @ out_w + out_b) ----------------
__global__ __launch_bounds__(256) void finalize_kernel(
    const float* __restrict__ moe, const float* __restrict__ spec,
    const float* __restrict__ out_w, const float* __restrict__ out_b,
    const float* __restrict__ bk_scale, float* __restrict__ out)
{
    size_t i = (size_t)blockIdx.x * blockDim.x + threadIdx.x;
    if (i >= (size_t)TTOK * TD) return;
    int d = (int)(i & (TD - 1));
    size_t t = i >> 9;
    float gr = spec[t * 2], gi = spec[t * 2 + 1];
    float so = gr * out_w[d] + gi * out_w[TD + d] + out_b[d];
    out[i] = moe[i] + bk_scale[0] * so;
}

extern "C" void kernel_launch(void* const* d_in, const int* in_sizes, int n_in,
                              void* d_out, int out_size, void* d_ws, size_t ws_size,
                              hipStream_t stream) {
    const float* x        = (const float*)d_in[0];
    const float* ln_gamma = (const float*)d_in[1];
    const float* ln_beta  = (const float*)d_in[2];
    const float* gate_w   = (const float*)d_in[3];
    const float* gate_b   = (const float*)d_in[4];
    const float* w1       = (const float*)d_in[5];
    const float* b1       = (const float*)d_in[6];
    const float* w2       = (const float*)d_in[7];
    const float* b2       = (const float*)d_in[8];
    const float* proj_w   = (const float*)d_in[9];
    const float* proj_b   = (const float*)d_in[10];
    const float* out_w    = (const float*)d_in[11];
    const float* out_b    = (const float*)d_in[12];
    const float* bk_scale = (const float*)d_in[13];

    char* ws = (char*)d_ws;
    __bf16* w1p  = (__bf16*)(ws + OFF_W1P);
    __bf16* w2p  = (__bf16*)(ws + OFF_W2P);
    __bf16* xn   = (__bf16*)(ws + OFF_XN);
    float*  moe  = (float*)(ws + OFF_MOE);
    float*  tp   = (float*)(ws + OFF_TP);
    float*  vbuf = (float*)(ws + OFF_V);
    float*  ubuf = (float*)(ws + OFF_U);
    float*  spec = (float*)(ws + OFF_SPEC);
    int*    list = (int*)(ws + OFF_LIST);
    int*    cnt  = (int*)(ws + OFF_CNT);

    float*  out  = (float*)d_out;

    zero_counts_kernel<<<1, 32, 0, stream>>>(cnt);
    pack_w1_kernel<<<(TE * TD * TH + 255) / 256, 256, 0, stream>>>(w1, w1p);
    pack_w2_kernel<<<(TE * TH * TD + 255) / 256, 256, 0, stream>>>(w2, w2p);
    ln_gate_kernel<<<TTOK / 8, 256, 0, stream>>>(x, ln_gamma, ln_beta, gate_w, gate_b,
                                                 xn, tp, list, cnt);
    dim3 ggrid(TTOK / 16, TE);
    moe_gemm_kernel<<<ggrid, 256, 32896, stream>>>(xn, w1p, w2p, b1, b2, tp, list, cnt, moe);
    proj_v_kernel<<<TTOK / 8, 256, 0, stream>>>(moe, proj_w, proj_b, vbuf);
    green_scan_kernel<<<1, 32, 0, stream>>>(vbuf, ubuf, spec);
    finalize_kernel<<<((size_t)TTOK * TD + 255) / 256, 256, 0, stream>>>(
        moe, spec, out_w, out_b, bk_scale, out);
}
